// Model_57028575756779
// MI455X (gfx1250) — matrix operand order repaired and checked by reading, not compiled, not run on hardware
//
#include <hip/hip_runtime.h>
#include <math.h>

typedef __attribute__((ext_vector_type(2)))  float        v2f;
typedef __attribute__((ext_vector_type(8)))  float        v8f;
typedef __attribute__((ext_vector_type(16))) __bf16       v16bf;
typedef __attribute__((ext_vector_type(4)))  unsigned int u32x4;
typedef __attribute__((ext_vector_type(4)))  int          i32x4;
typedef __attribute__((ext_vector_type(8)))  int          i32x8;

#define B_    128
#define RES_  512
#define WIN_  32
#define NSEG_ 64
#define WASH_ 4
#define KRO_  30720   // (64-4)*512
#define ORO_  8192    // 16*512
#define SEQ_  2048

#define TROW_ 36      // padded LDS row stride (32 data + 4 pad DWORDs, bank-conflict free)

// ---- helpers -------------------------------------------------------------

__device__ __forceinline__ v16bf cvt16(float4 a, float4 b, float4 c, float4 d) {
    v16bf r;
    r[0]  = (__bf16)a.x; r[1]  = (__bf16)a.y; r[2]  = (__bf16)a.z; r[3]  = (__bf16)a.w;
    r[4]  = (__bf16)b.x; r[5]  = (__bf16)b.y; r[6]  = (__bf16)b.z; r[7]  = (__bf16)b.w;
    r[8]  = (__bf16)c.x; r[9]  = (__bf16)c.y; r[10] = (__bf16)c.z; r[11] = (__bf16)c.w;
    r[12] = (__bf16)d.x; r[13] = (__bf16)d.y; r[14] = (__bf16)d.z; r[15] = (__bf16)d.w;
    return r;
}

// Issue one TDM 2D tile load: 128 rows x 32 fp32 (one k-chunk of H) -> LDS,
// with 4-DWORD padding after every 32 DWORDs (LDS row stride = 36 floats).
// D# packing per CDNA5 ISA.
// This toolchain declares the 6-arg builtin:
//   (u32x4 g0, i32x8 g1, i32x4 g2, i32x4 g3, i32x8 extra, i32 cpol)
__device__ __forceinline__ void tdm_load_chunk(unsigned long long gaddr, unsigned ldsaddr)
{
    const unsigned glo = (unsigned)__builtin_amdgcn_readfirstlane((int)(gaddr & 0xFFFFFFFFull));
    const unsigned ghi = (unsigned)__builtin_amdgcn_readfirstlane((int)(gaddr >> 32));

    u32x4 g0;
    g0[0] = 1u;                                   // count=1 (valid), user mode, no gather
    g0[1] = ldsaddr;                              // lds_addr (bytes)
    g0[2] = glo;                                  // global_addr[31:0]
    g0[3] = (ghi & 0x01FFFFFFu) | (2u << 30);     // global_addr[56:32] | type=2 (image)

    i32x8 g1;
    g1[0] = 0x07120000;   // data_size=4B | pad_enable | pad_interval=32dw | pad_amount=4dw
    g1[1] = 0x78000000;   // tensor_dim0 = 30720 (low16 in bits [63:48])
    g1[2] = 0x00800000;   // tensor_dim0 hi = 0; tensor_dim1 = 128 (low16 at [95:80])
    g1[3] = 0x00200000;   // tensor_dim1 hi = 0; tile_dim0 = 32
    g1[4] = 0x00000080;   // tile_dim1 = 128; tile_dim2 = 0
    g1[5] = 0x00007800;   // tensor_dim0_stride = 30720 (low 32)
    g1[6] = 0;            // stride0 hi; tensor_dim1_stride low (unused, 2D tile)
    g1[7] = 0;

    const i32x4 z4 = {0, 0, 0, 0};                // groups 2/3: tensor is 2D
    const i32x8 z8 = {0, 0, 0, 0, 0, 0, 0, 0};
    __builtin_amdgcn_tensor_load_to_lds(g0, g1, z4, z4, z8, 0);
}

// ---- kernel 1: one ESN recurrence step (fp32 WMMA, launched 64x) ---------
// h_t[b, n] = tanh( sum_k x_t[b,k] Win[n,k] + sum_k h_{t-1}[b,k] Wres[n,k] )
// One wave computes a 16(batch) x 16(res) tile. grid = 8 x 32 tiles.

__global__ __launch_bounds__(128)
void esn_step(const float* __restrict__ x, const float* __restrict__ Win,
              const float* __restrict__ Wres, const float* __restrict__ hprev,
              float* __restrict__ hcur, float* __restrict__ hcat, int t)
{
    const int lane  = threadIdx.x & 31;
    const int wave  = blockIdx.x * (blockDim.x >> 5) + (threadIdx.x >> 5);
    const int n0    = (wave & 31) * 16;   // reservoir-unit tile
    const int b0    = (wave >> 5) * 16;   // batch tile
    const int r15   = lane & 15;          // A row / B column within tile
    const int half  = lane >> 4;          // K-half select

    v8f acc = {};

    // ---- x_t @ Win^T : K = 32 ----
    const float* xrow  = x   + (size_t)(b0 + r15) * SEQ_ + (size_t)t * WIN_;
    const float* wirow = Win + (size_t)(n0 + r15) * WIN_;
    #pragma unroll
    for (int kc = 0; kc < WIN_ / 4; ++kc) {
        const int k0 = kc * 4 + half * 2;
        v2f a = *(const v2f*)(xrow  + k0);   // A: 16x4 fp32, row-per-lane
        v2f b = *(const v2f*)(wirow + k0);   // B: 4x16 fp32, col-per-lane
        acc = __builtin_amdgcn_wmma_f32_16x16x4_f32(false, a, false, b,
                                                    (short)0, acc, false, false);
    }

    // ---- h_{t-1} @ Wres^T : K = 512 ----
    const float* hrow  = hprev + (size_t)(b0 + r15) * RES_;
    const float* wrrow = Wres  + (size_t)(n0 + r15) * RES_;
    #pragma unroll 4
    for (int kc = 0; kc < RES_ / 4; ++kc) {
        const int k0 = kc * 4 + half * 2;
        v2f a = *(const v2f*)(hrow  + k0);
        v2f b = *(const v2f*)(wrrow + k0);
        acc = __builtin_amdgcn_wmma_f32_16x16x4_f32(false, a, false, b,
                                                    (short)0, acc, false, false);
    }

    // ---- tanh + store (D layout: col = lane&15, rows = half*8 + i) ----
    const int n = n0 + r15;
    #pragma unroll
    for (int i = 0; i < 8; ++i) {
        const int b = b0 + half * 8 + i;
        const float v = tanhf(acc[i]);
        hcur[(size_t)b * RES_ + n] = v;
        if (t >= WASH_)
            hcat[(size_t)b * KRO_ + (size_t)(t - WASH_) * RES_ + n] = v;
    }
}

// ---- kernel 2: readout GEMM  Y[128,8192] = H[128,30720] @ W^T + bias -----
// Bandwidth-bound on the 1 GB fp32 weight stream (read exactly once: each
// block owns 128 output columns). The shared 128x32 A-chunk is DMA'd to LDS
// by the Tensor Data Mover (double-buffered, TENSORcnt), and all 8 waves of
// the block feed bf16 WMMAs from conflict-free padded LDS rows.

__global__ __launch_bounds__(256)
void readout(const float* __restrict__ H, const float* __restrict__ W,
             const float* __restrict__ bias, float* __restrict__ Y)
{
    __shared__ __align__(16) float tileA[2][B_ * TROW_];   // 2 x 18 KB

    const int lane = threadIdx.x & 31;
    const int wvid = threadIdx.x >> 5;                 // 0..7
    const int wave = blockIdx.x * 8 + wvid;            // 0..511
    const int o0   = wave * 16;
    const int c15  = lane & 15;
    const int half = lane >> 4;

    const unsigned long long hbase = (unsigned long long)(const void*)H;
    const unsigned lds0 = (unsigned)(size_t)(&tileA[0][0]);
    const unsigned lds1 = (unsigned)(size_t)(&tileA[1][0]);

    const int NCH = KRO_ / 32;                         // 960 k-chunks

    if (wvid == 0)
        tdm_load_chunk(hbase, lds0);                   // chunk 0 -> buf 0

    v8f acc[8] = {};
    const float* wp = W + (size_t)(o0 + c15) * KRO_ + half * 16;

    for (int ch = 0; ch < NCH; ++ch) {
        if (wvid == 0)
            __builtin_amdgcn_s_wait_tensorcnt(0);      // chunk ch resident
        __syncthreads();                               // publish buf; prev buf fully consumed
        if (wvid == 0 && ch + 1 < NCH)                 // overlap DMA of ch+1 with compute
            tdm_load_chunk(hbase + (unsigned long long)(ch + 1) * 32u * 4u,
                           (ch & 1) ? lds0 : lds1);

        // B fragment: 16 consecutive fp32 of one readout_w row -> bf16
        float4 w0 = *(const float4*)(wp + 0);
        float4 w1 = *(const float4*)(wp + 4);
        float4 w2 = *(const float4*)(wp + 8);
        float4 w3 = *(const float4*)(wp + 12);
        __builtin_prefetch(wp + 512, 0, 1);            // global_prefetch ahead on W stream
        const v16bf bf = cvt16(w0, w1, w2, w3);
        wp += 32;

        const float* la = &tileA[ch & 1][0];
        #pragma unroll
        for (int m = 0; m < 8; ++m) {                  // full batch: 8 M-tiles
            const float* ap = la + (m * 16 + c15) * TROW_ + half * 16;
            float4 a0 = *(const float4*)(ap + 0);      // ds_load_b128 x4, padded rows
            float4 a1 = *(const float4*)(ap + 4);
            float4 a2 = *(const float4*)(ap + 8);
            float4 a3 = *(const float4*)(ap + 12);
            acc[m] = __builtin_amdgcn_wmma_f32_16x16x32_bf16(false, cvt16(a0, a1, a2, a3),
                                                             false, bf, (short)0, acc[m],
                                                             false, false);
        }
    }

    const float bv = bias[o0 + c15];
    #pragma unroll
    for (int m = 0; m < 8; ++m) {
        #pragma unroll
        for (int i = 0; i < 8; ++i) {
            const int r = m * 16 + half * 8 + i;
            Y[(size_t)r * ORO_ + o0 + c15] = acc[m][i] + bv;
        }
    }
}

// ---- kernel 3: projection  out[b, seg*32+w] = Y[b,seg,:] . proj_w[w,:] ---

__global__ __launch_bounds__(256)
void proj(const float* __restrict__ Y, const float* __restrict__ Pw,
          const float* __restrict__ Pb, float* __restrict__ out)
{
    const int idx = blockIdx.x * blockDim.x + threadIdx.x;  // 0..65535
    const int b   = idx >> 9;
    const int p   = idx & 511;
    const int seg = p >> 5;
    const int w   = p & 31;
    const float* yrow = Y  + (size_t)b * ORO_ + (size_t)seg * RES_;
    const float* wrow = Pw + (size_t)w * RES_;
    float s = Pb[w];
    #pragma unroll 4
    for (int r = 0; r < RES_; r += 4) {
        float4 yv = *(const float4*)(yrow + r);
        float4 wv = *(const float4*)(wrow + r);
        s += yv.x * wv.x + yv.y * wv.y + yv.z * wv.z + yv.w * wv.w;
    }
    out[idx] = s;   // output is [B,512,1] flat = b*512 + p
}

// ---- init ----------------------------------------------------------------

__global__ void zero_f32(float* p, int n)
{
    const int i = blockIdx.x * blockDim.x + threadIdx.x;
    if (i < n) p[i] = 0.0f;
}

// ---- launch --------------------------------------------------------------

extern "C" void kernel_launch(void* const* d_in, const int* in_sizes, int n_in,
                              void* d_out, int out_size, void* d_ws, size_t ws_size,
                              hipStream_t stream)
{
    (void)in_sizes; (void)n_in; (void)out_size; (void)ws_size;

    const float* x    = (const float*)d_in[0];  // [128,2048,1]
    const float* Win  = (const float*)d_in[1];  // [512,32]
    const float* Wres = (const float*)d_in[2];  // [512,512]
    const float* Row  = (const float*)d_in[3];  // [8192,30720]
    const float* Rob  = (const float*)d_in[4];  // [8192]
    const float* Pw   = (const float*)d_in[5];  // [32,512]
    const float* Pb   = (const float*)d_in[6];  // [32]
    float* out = (float*)d_out;

    float* ws    = (float*)d_ws;
    float* hping = ws;                               // 2 * 128 * 512
    float* hcat  = ws + 2 * B_ * RES_;               // 128 * 30720
    float* Y     = hcat + (size_t)B_ * KRO_;         // 128 * 8192

    // h0 = 0 (workspace is not re-poisoned between replays -> zero each call)
    zero_f32<<<(2 * B_ * RES_ + 255) / 256, 256, 0, stream>>>(hping, 2 * B_ * RES_);

    // 64 sequential recurrence steps (stream order gives the dependency chain)
    for (int t = 0; t < NSEG_; ++t) {
        const float* hp = hping + (size_t)(t & 1) * (B_ * RES_);
        float*       hc = hping + (size_t)((t + 1) & 1) * (B_ * RES_);
        esn_step<<<64, 128, 0, stream>>>(x, Win, Wres, hp, hc, hcat, t);
    }

    // big bandwidth-bound GEMM (TDM-staged A, bf16 WMMA), then tiny projection
    readout<<<64, 256, 0, stream>>>(hcat, Row, Rob, Y);
    proj<<<(B_ * 512) / 256, 256, 0, stream>>>(Y, Pw, Pb, out);
}
